// GCCaps_6674379178205
// MI455X (gfx1250) — matrix operand order repaired and checked by reading, not compiled, not run on hardware
//
#include <hip/hip_runtime.h>
#include <hip/hip_bf16.h>
#include <math.h>

// ---------------------------------------------------------------------------
// Types for WMMA fragments (CDNA5 / gfx1250, wave32)
// ---------------------------------------------------------------------------
typedef __attribute__((ext_vector_type(16))) _Float16 v16h;
typedef __attribute__((ext_vector_type(8)))  float    v8f;
typedef __attribute__((ext_vector_type(4))) unsigned int u32x4;
typedef __attribute__((ext_vector_type(8)))  int      i32x8;
typedef __attribute__((ext_vector_type(4)))  int      i32x4;

union Frag16 { v16h v; u32x4 q[2]; };     // 32 bytes: one A or B fp16 fragment per lane
union Pack8  { _Float16 h[8]; u32x4 q; }; // 8 fp16 -> one 16B store

__device__ __forceinline__ float sigmoidf_(float x) {
    return 1.0f / (1.0f + __expf(-x));
}

// ---------------------------------------------------------------------------
// Kernel 0: repack conv weights OIHW fp32 -> [O][tap][I] fp16 (im2col K-order)
// ---------------------------------------------------------------------------
__global__ void convert_w_kernel(const float* __restrict__ w, _Float16* __restrict__ out) {
    int tid = blockIdx.x * blockDim.x + threadIdx.x;
    if (tid >= 64 * 9 * 64) return;
    int i   = tid & 63;          // input channel
    int tap = (tid >> 6) % 9;    // ky*3+kx
    int o   = tid / (64 * 9);    // output channel
    out[(o * 9 + tap) * 64 + i] = (_Float16)w[(o * 64 + i) * 9 + tap];
}

// ---------------------------------------------------------------------------
// Kernel 1: cram1 = gated conv (1->64, 3x3, pad 1) + 2x2 maxpool.
// Input x (32,256,128) fp32 -> h1 (B=32,H=128,W=64,C=64) channels-last fp16.
// ---------------------------------------------------------------------------
__global__ __launch_bounds__(256)
void cram1_kernel(const float* __restrict__ x,
                  const float* __restrict__ wl, const float* __restrict__ bl,
                  const float* __restrict__ wg, const float* __restrict__ bg,
                  _Float16* __restrict__ out) {
    int tid = blockIdx.x * blockDim.x + threadIdx.x;
    if (tid >= 32 * 128 * 64 * 64) return;
    int c  = tid & 63;
    int xo = (tid >> 6) & 63;
    int yo = (tid >> 12) & 127;
    int b  = tid >> 19;

    float WL[9], WG[9];
#pragma unroll
    for (int t = 0; t < 9; ++t) { WL[t] = wl[c * 9 + t]; WG[t] = wg[c * 9 + t]; }

    const float* xb = x + (size_t)b * 256 * 128;
    float best = -3.4e38f;
#pragma unroll
    for (int dy = 0; dy < 2; ++dy) {
#pragma unroll
        for (int dx = 0; dx < 2; ++dx) {
            int Y = 2 * yo + dy, X = 2 * xo + dx;
            float sl = bl[c], sg = bg[c];
#pragma unroll
            for (int ky = 0; ky < 3; ++ky) {
                int iy = Y + ky - 1;
                if (iy < 0 || iy >= 256) continue;
#pragma unroll
                for (int kx = 0; kx < 3; ++kx) {
                    int ix = X + kx - 1;
                    if (ix < 0 || ix >= 128) continue;
                    float v = xb[iy * 128 + ix];
                    sl = fmaf(v, WL[ky * 3 + kx], sl);
                    sg = fmaf(v, WG[ky * 3 + kx], sg);
                }
            }
            best = fmaxf(best, sl * sigmoidf_(sg));
        }
    }
    out[((((size_t)b * 128 + yo) * 64 + xo) << 6) + c] = (_Float16)best;
}

// ---------------------------------------------------------------------------
// TDM helper: 1-D tensor_load_to_lds of `nelem` fp16 elements (nelem < 65536,
// byte size multiple of 4) from `gsrc` into LDS byte address `lds_addr`.
// Descriptor packing per CDNA5 ISA section 8.3/8.4. This toolchain exposes
// the 6-arg builtin: (g0, g1, g2, g3, g_extra, cpol).
// ---------------------------------------------------------------------------
__device__ __forceinline__ void tdm_load_fp16_1d(const _Float16* gsrc,
                                                 unsigned lds_addr,
                                                 unsigned nelem) {
    unsigned long long ga = (unsigned long long)(uintptr_t)gsrc;
    u32x4 g0;
    g0[0] = 1u;                                   // count=1 (valid user D#)
    g0[1] = lds_addr;                             // lds_addr (bits 63:32)
    g0[2] = (unsigned)(ga & 0xFFFFFFFFu);         // global_addr[31:0]
    g0[3] = (unsigned)((ga >> 32) & 0x01FFFFFFu)  // global_addr[56:32]
            | (2u << 30);                         // type = 2 ("image")
    i32x8 g1;
    g1[0] = 0x00010000;                           // data_size=1 (2 bytes/elem)
    g1[1] = (int)((nelem & 0xFFFFu) << 16);       // tensor_dim0[15:0]
    g1[2] = (int)((nelem >> 16) | (1u << 16));    // tensor_dim0[31:16] | tensor_dim1=1
    g1[3] = (int)((nelem & 0xFFFFu) << 16);       // tile_dim0 = nelem
    g1[4] = 0;                                    // tile_dim1=0, tile_dim2=0 (1-D)
    g1[5] = (int)nelem;                           // tensor_dim0_stride[31:0]
    g1[6] = 0;
    g1[7] = 0;
    i32x4 gz4 = {0, 0, 0, 0};
    i32x8 gz8 = {0, 0, 0, 0, 0, 0, 0, 0};
    __builtin_amdgcn_tensor_load_to_lds(g0, g1, gz4, gz4, gz8, 0);
}

// ---------------------------------------------------------------------------
// Kernel 2: WMMA implicit-GEMM gated conv (64->64, 3x3, pad 1) + 2x2 maxpool.
//
// Weights (both streams, [O][tap][I] fp16, 2 x 73,728 B) are staged into LDS
// once per workgroup by the Tensor Data Mover; the K-loop then feeds the A
// fragments from LDS (ds_load) and only the activation B fragment touches
// global memory (L2-resident). One wave owns a 16-pixel (2x8) x 64-channel
// tile for lin+gate; K = 18 chunks of 32 -> 8 WMMAs per chunk.
// Fragment lane layouts per ISA 7.12.2; 2x2 maxpool via __shfl_xor(1)/(8).
// ---------------------------------------------------------------------------
template <int H, int W>
__global__ __launch_bounds__(128)
void cram_gemm_kernel(const _Float16* __restrict__ in,
                      const _Float16* __restrict__ wl, const _Float16* __restrict__ wg,
                      const float* __restrict__ bl, const float* __restrict__ bg,
                      _Float16* __restrict__ out) {
    extern __shared__ _Float16 wsm[];            // [2][36864] = 147,456 B
    const int lane = threadIdx.x & 31;
    const int wave = threadIdx.x >> 5;

    // ---- stage weights into LDS with the TDM (wave 0 issues, all wait) ----
    if (wave == 0) {
        const unsigned lds_base = (unsigned)(uintptr_t)(void*)wsm;
        tdm_load_fp16_1d(wl, lds_base, 36864u);
        tdm_load_fp16_1d(wg, lds_base + 73728u, 36864u);
        __builtin_amdgcn_s_wait_tensorcnt(0);
    }
    __syncthreads();
    const _Float16* __restrict__ wsl = wsm;
    const _Float16* __restrict__ wsg = wsm + 36864;

    const int c = blockIdx.x * 4 + wave;   // column tile: 8 pre-pool cols
    const int r = blockIdx.y;              // row pair: pre-pool rows 2r,2r+1
    const int b = blockIdx.z;
    if (c * 8 >= W) return;                // wave-uniform guard

    const int n  = lane & 15;              // pixel index within 16-wide N tile
    const int hi = lane >> 4;              // lane half (K/M split)
    const int py = 2 * r + (n >> 3);
    const int px = c * 8 + (n & 7);
    const int kb  = hi * 8;                // B-fragment K base for this lane
    const int klo = hi * 8;                // A-fragment K base for this lane

    v8f accL[4] = {};
    v8f accG[4] = {};
    const u32x4 zero4 = {0u, 0u, 0u, 0u};

    for (int kc = 0; kc < 18; ++kc) {
        const int tap = kc >> 1;
        const int ch0 = (kc & 1) * 32;
        const int ky = tap / 3, kx = tap % 3;

        // ---- B fragment: im2col activations (channels-last, contiguous) ---
        Frag16 bf;
        const int iy = py + ky - 1;
        const int ix = px + kx - 1;
        if (iy >= 0 && iy < H && ix >= 0 && ix < W) {
            const _Float16* src =
                in + (((size_t)(b * H + iy) * W + ix) << 6) + ch0 + kb;
            bf.q[0] = *(const u32x4*)(src);
            bf.q[1] = *(const u32x4*)(src + 16);
        } else {
            bf.q[0] = zero4;
            bf.q[1] = zero4;
        }

        // ---- prefetch next K-chunk's activation line (global_prefetch) ----
        if (kc + 1 < 18) {
            const int tap2 = (kc + 1) >> 1;
            const int iy2 = py + tap2 / 3 - 1;
            const int ix2 = px + tap2 % 3 - 1;
            if (iy2 >= 0 && iy2 < H && ix2 >= 0 && ix2 < W) {
                const _Float16* nsrc =
                    in + (((size_t)(b * H + iy2) * W + ix2) << 6);
                __builtin_prefetch((const void*)nsrc, 0, 0);
            }
        }

        // ---- A fragments from LDS: gather all 8 first, then 8 WMMAs -------
        Frag16 al[4], ag[4];
#pragma unroll
        for (int mt = 0; mt < 4; ++mt) {
            const int o = mt * 16 + n;     // output channel = M row
            const int wo = ((o * 9 + tap) << 6) + ch0 + klo;
            al[mt].q[0] = *(const u32x4*)(wsl + wo);
            al[mt].q[1] = *(const u32x4*)(wsl + wo + 16);
            ag[mt].q[0] = *(const u32x4*)(wsg + wo);
            ag[mt].q[1] = *(const u32x4*)(wsg + wo + 16);
        }
#pragma unroll
        for (int mt = 0; mt < 4; ++mt) {
            accL[mt] = __builtin_amdgcn_wmma_f32_16x16x32_f16(
                false, al[mt].v, false, bf.v, (short)0, accL[mt], false, false);
            accG[mt] = __builtin_amdgcn_wmma_f32_16x16x32_f16(
                false, ag[mt].v, false, bf.v, (short)0, accG[mt], false, false);
        }
    }

    // ---- epilogue: bias + sigmoid gate + 2x2 maxpool (shfl) + fp16 store ---
    const int q = n >> 1;                          // pooled column within tile
    const bool writer = ((n & 1) == 0) && (n < 8); // row 0, even col
    _Float16* outp =
        out + ((((size_t)b * (H / 2) + r) * (W / 2)) + (c * 4 + q)) * 64;

#pragma unroll
    for (int mt = 0; mt < 4; ++mt) {
        const int mbase = mt * 16 + hi * 8;
        Pack8 pk;
#pragma unroll
        for (int j = 0; j < 8; ++j) {
            float lv = accL[mt][j] + bl[mbase + j];
            float gv = accG[mt][j] + bg[mbase + j];
            float g  = lv * sigmoidf_(gv);
            g = fmaxf(g, __shfl_xor(g, 1));  // pool across col pair
            g = fmaxf(g, __shfl_xor(g, 8));  // pool across row pair
            pk.h[j] = (_Float16)g;
        }
        if (writer) *(u32x4*)(outp + mbase) = pk.q;
    }
}

// ---------------------------------------------------------------------------
// Kernel 3: primary-caps conv (64->64, 3x3, pad 1, stride (1,2)) on h3.
// Writes pre-squash capsule tensor s in (B, T=32, I=128, 4) layout.
// ---------------------------------------------------------------------------
__global__ __launch_bounds__(256)
void prim_kernel(const _Float16* __restrict__ h3, const float* __restrict__ w,
                 const float* __restrict__ bias, float* __restrict__ s) {
    int tid = blockIdx.x * blockDim.x + threadIdx.x;
    if (tid >= 32 * 32 * 8 * 64) return;
    int co = tid & 63;
    int f  = (tid >> 6) & 7;
    int t  = (tid >> 9) & 31;
    int b  = tid >> 14;

    float acc = bias[co];
#pragma unroll
    for (int ky = 0; ky < 3; ++ky) {
        int iy = t + ky - 1;
        if (iy < 0 || iy >= 32) continue;
#pragma unroll
        for (int kx = 0; kx < 3; ++kx) {
            int ix = 2 * f + kx - 1;
            if (ix < 0 || ix >= 16) continue;
            const _Float16* ip = h3 + (((size_t)(b * 32 + iy) * 16 + ix) << 6);
            const int tapo = ky * 3 + kx;
            for (int ci = 0; ci < 64; ++ci)
                acc = fmaf((float)ip[ci], w[(co * 64 + ci) * 9 + tapo], acc);
        }
    }
    s[((size_t)(b * 32 + t) * 128 + (f * 16 + (co >> 2))) * 4 + (co & 3)] = acc;
}

// ---------------------------------------------------------------------------
// Kernel 4: zero the output (harness poisons d_out; routing accumulates).
// ---------------------------------------------------------------------------
__global__ void zero_out_kernel(float* out) {
    if (threadIdx.x < 320) out[threadIdx.x] = 0.0f;
}

// ---------------------------------------------------------------------------
// Kernel 5: dynamic routing. One workgroup per (b, t); u_hat (40KB) and all
// routing state live in LDS (~53KB << 320KB/WGP).
// ---------------------------------------------------------------------------
__global__ __launch_bounds__(256)
void routing_kernel(const float* __restrict__ s, const float* __restrict__ Wc,
                    float* __restrict__ out) {
    __shared__ float u_s[128][4];
    __shared__ float uh_s[10][128][8];
    __shared__ float b_s[10][128];
    __shared__ float c_s[10][128];
    __shared__ float vr_s[10][8];
    __shared__ float v_s[10][8];

    const int tid = threadIdx.x;
    const int b = blockIdx.x >> 5;
    const int t = blockIdx.x & 31;
    const float* sp = s + (size_t)(b * 32 + t) * 512;

    for (int idx = tid; idx < 512; idx += 256) u_s[idx >> 2][idx & 3] = sp[idx];
    for (int idx = tid; idx < 1280; idx += 256) b_s[idx >> 7][idx & 127] = 0.0f;
    __syncthreads();

    if (tid < 128) {
        float n2 = 0.0f;
        for (int k = 0; k < 4; ++k) n2 += u_s[tid][k] * u_s[tid][k];
        float sc = (n2 / (1.0f + n2)) * rsqrtf(n2 + 1e-8f);
        for (int k = 0; k < 4; ++k) u_s[tid][k] *= sc;
    }
    __syncthreads();

    for (int idx = tid; idx < 10240; idx += 256) {
        int j = idx >> 10;
        int i = (idx >> 3) & 127;
        int m = idx & 7;
        const float* wp = Wc + (((j * 128 + i) * 8 + m) << 2);
        float acc = 0.0f;
        for (int k = 0; k < 4; ++k) acc = fmaf(wp[k], u_s[i][k], acc);
        uh_s[j][i][m] = acc;
    }
    __syncthreads();

    for (int it = 0; it < 3; ++it) {
        if (tid < 128) {
            float mx = b_s[0][tid];
            for (int j = 1; j < 10; ++j) mx = fmaxf(mx, b_s[j][tid]);
            float e[10], sum = 0.0f;
            for (int j = 0; j < 10; ++j) { e[j] = __expf(b_s[j][tid] - mx); sum += e[j]; }
            float inv = 1.0f / sum;
            for (int j = 0; j < 10; ++j) c_s[j][tid] = e[j] * inv;
        }
        __syncthreads();
        if (tid < 80) {
            int j = tid >> 3, m = tid & 7;
            float acc = 0.0f;
            for (int i = 0; i < 128; ++i) acc = fmaf(c_s[j][i], uh_s[j][i][m], acc);
            vr_s[j][m] = acc;
        }
        __syncthreads();
        if (tid < 10) {
            float n2 = 0.0f;
            for (int m = 0; m < 8; ++m) n2 += vr_s[tid][m] * vr_s[tid][m];
            float sc = (n2 / (1.0f + n2)) * rsqrtf(n2 + 1e-8f);
            for (int m = 0; m < 8; ++m) v_s[tid][m] = vr_s[tid][m] * sc;
        }
        __syncthreads();
        for (int idx = tid; idx < 1280; idx += 256) {
            int j = idx >> 7, i = idx & 127;
            float acc = 0.0f;
            for (int m = 0; m < 8; ++m) acc = fmaf(v_s[j][m], uh_s[j][i][m], acc);
            b_s[j][i] += acc;
        }
        __syncthreads();
    }

    if (tid < 10) {
        float n2 = 1e-8f;
        for (int m = 0; m < 8; ++m) n2 += v_s[tid][m] * v_s[tid][m];
        atomicAdd(&out[b * 10 + tid], sqrtf(n2) * (1.0f / 32.0f));
    }
}

// ---------------------------------------------------------------------------
// Launch
// ---------------------------------------------------------------------------
extern "C" void kernel_launch(void* const* d_in, const int* in_sizes, int n_in,
                              void* d_out, int out_size, void* d_ws, size_t ws_size,
                              hipStream_t stream) {
    (void)in_sizes; (void)n_in; (void)out_size; (void)ws_size;

    const float* x       = (const float*)d_in[0];
    const float* w_lin1  = (const float*)d_in[1];
    const float* b_lin1  = (const float*)d_in[2];
    const float* w_gate1 = (const float*)d_in[3];
    const float* b_gate1 = (const float*)d_in[4];
    const float* w_lin2  = (const float*)d_in[5];
    const float* b_lin2  = (const float*)d_in[6];
    const float* w_gate2 = (const float*)d_in[7];
    const float* b_gate2 = (const float*)d_in[8];
    const float* w_lin3  = (const float*)d_in[9];
    const float* b_lin3  = (const float*)d_in[10];
    const float* w_gate3 = (const float*)d_in[11];
    const float* b_gate3 = (const float*)d_in[12];
    const float* w_prim  = (const float*)d_in[13];
    const float* b_prim  = (const float*)d_in[14];
    const float* W_caps  = (const float*)d_in[15];
    float* out = (float*)d_out;

    // Workspace layout (bytes)
    char* ws = (char*)d_ws;
    _Float16* h1  = (_Float16*)(ws);                         // 33,554,432 B
    _Float16* h2  = (_Float16*)(ws + 33554432);              //  8,388,608 B
    _Float16* h3  = (_Float16*)(ws + 33554432 + 8388608);    //  2,097,152 B
    _Float16* wl2 = (_Float16*)(ws + 44040192);              //  4 x 73,728 B
    _Float16* wg2 = wl2 + 36864;
    _Float16* wl3 = wg2 + 36864;
    _Float16* wg3 = wl3 + 36864;
    float*    sbuf = (float*)(ws + 44040192 + 294912);       //  2,097,152 B

    // Stage 0: pack fp16 weights in im2col K-order
    convert_w_kernel<<<144, 256, 0, stream>>>(w_lin2, wl2);
    convert_w_kernel<<<144, 256, 0, stream>>>(w_gate2, wg2);
    convert_w_kernel<<<144, 256, 0, stream>>>(w_lin3, wl3);
    convert_w_kernel<<<144, 256, 0, stream>>>(w_gate3, wg3);

    // Stage 1: cram1 (scalar; K=9)
    cram1_kernel<<<65536, 256, 0, stream>>>(x, w_lin1, b_lin1, w_gate1, b_gate1, h1);

    // Stage 2/3: WMMA gated conv + pool (147,456 B dynamic LDS for weights)
    cram_gemm_kernel<128, 64><<<dim3(2, 64, 32), 128, 147456, stream>>>(
        h1, wl2, wg2, b_lin2, b_gate2, h2);
    cram_gemm_kernel<64, 32><<<dim3(1, 32, 32), 128, 147456, stream>>>(
        h2, wl3, wg3, b_lin3, b_gate3, h3);

    // Stage 4: primary caps conv (stride (1,2)) -> pre-squash capsules
    prim_kernel<<<2048, 256, 0, stream>>>(h3, w_prim, b_prim, sbuf);

    // Stage 5: routing + length mean
    zero_out_kernel<<<1, 320, 0, stream>>>(out);
    routing_kernel<<<1024, 256, 0, stream>>>(sbuf, W_caps, out);
}
